// WeightOnlyInt8Linear_11269994185154
// MI455X (gfx1250) — matrix operand order repaired and checked by reading, not compiled, not run on hardware
//
#include <hip/hip_runtime.h>
#include <hip/hip_bf16.h>
#include <stdint.h>

// GEMM: out[M,N] = (x[M,K] fp32) x (w[N,K] int8 -> f16)^T * scale[N]
// M = B*S = 8192, K = 4096, N = 11008

typedef __attribute__((ext_vector_type(16))) _Float16 v16h;
typedef __attribute__((ext_vector_type(8)))  _Float16 v8h;
typedef __attribute__((ext_vector_type(2)))  _Float16 v2h;
typedef __attribute__((ext_vector_type(8)))  float    v8f;
typedef __attribute__((ext_vector_type(4)))  float    v4f;

#define BM 128
#define BN 128
#define BK 32
#define LDH 40  // padded LDS row stride in halves (80 B: 20-bank stride, conflict-free b128)

// int8 -> f16 exact dequant, 2 values per call:
//   u = b ^ 0x80  (bias into [0,255])
//   0x64'u' as f16 == 1024 + u  (u lands in the low 8 mantissa bits, 1 ulp = 1.0)
//   subtract 1152 (= 1024 + 128)  ->  b, exactly
__device__ __forceinline__ v2h dq2(uint32_t ubiased, uint32_t sel) {
    uint32_t w = __builtin_amdgcn_perm(0x64646464u, ubiased, sel);  // v_perm_b32
    v2h h = __builtin_bit_cast(v2h, w);
    return h - (v2h)((_Float16)1152.0f);                            // v_pk_add_f16
}

// dequant 16 int8 (int4 register) -> two v8h
__device__ __forceinline__ void dq16(const int4& bR, v8h& lo, v8h& hi) {
    const uint32_t* pd = (const uint32_t*)&bR;
#pragma unroll
    for (int i = 0; i < 2; ++i) {
        uint32_t u0 = pd[2 * i + 0] ^ 0x80808080u;   // bytes 8i+0..3
        uint32_t u1 = pd[2 * i + 1] ^ 0x80808080u;   // bytes 8i+4..7
        v2h a = dq2(u0, 0x04010400u);   // b0,b1
        v2h b = dq2(u0, 0x04030402u);   // b2,b3
        v2h c = dq2(u1, 0x04010400u);   // b4,b5
        v2h d = dq2(u1, 0x04030402u);   // b6,b7
        v8h& dst = (i == 0) ? lo : hi;
        dst[0] = a[0]; dst[1] = a[1]; dst[2] = b[0]; dst[3] = b[1];
        dst[4] = c[0]; dst[5] = c[1]; dst[6] = d[0]; dst[7] = d[1];
    }
}

__global__ __launch_bounds__(256)
void woq_int8_gemm_wmma(const float* __restrict__ X,
                        const int8_t* __restrict__ W,
                        const __hip_bfloat16* __restrict__ S,
                        float* __restrict__ Out,
                        int M, int N, int K)
{
    // double-buffered LDS: 2 * (10 KB A + 10 KB B) = 40 KB
    __shared__ __align__(16) _Float16 As[2][BM * LDH];
    __shared__ __align__(16) _Float16 Bs[2][BN * LDH];

    const int tid  = threadIdx.x;
    const int lane = tid & 31;
    const int wave = tid >> 5;       // 0..7
    const int wm   = wave >> 2;      // 0..1 : wave owns 64 rows of M
    const int wn   = wave & 3;       // 0..3 : wave owns 32 cols of N
    const int lm   = lane & 15;
    const int hi   = lane >> 4;      // half-wave select (ISA fragment layout)

    const int m0 = blockIdx.y * BM;
    const int n0 = blockIdx.x * BN;

    // staging map: each thread owns 16 elems of K in one row
    const int grow = tid >> 1;           // 0..127
    const int gcol = (tid & 1) * 16;     // 0 or 16

    const float*  gA = X + (size_t)(m0 + grow) * K + gcol;
    const int8_t* gB = W + (size_t)(n0 + grow) * K + gcol;

    v8f acc[4][2];
#pragma unroll
    for (int mt = 0; mt < 4; ++mt)
#pragma unroll
        for (int nt = 0; nt < 2; ++nt)
#pragma unroll
            for (int r = 0; r < 8; ++r) acc[mt][nt][r] = 0.0f;

    // staging registers (live across the WMMA block)
    v4f  aF0, aF1, aF2, aF3;
    int4 bR;

    // ---- prologue: load + convert + store chunk 0 into buffer 0 ------------
    aF0 = *(const v4f*)(gA + 0);
    aF1 = *(const v4f*)(gA + 4);
    aF2 = *(const v4f*)(gA + 8);
    aF3 = *(const v4f*)(gA + 12);
    bR  = *(const int4*)(gB);
    {
        _Float16* sA = &As[0][grow * LDH + gcol];
        _Float16* sB = &Bs[0][grow * LDH + gcol];
        v8h a0, a1, b0, b1;
#pragma unroll
        for (int j = 0; j < 4; ++j) {
            a0[j] = (_Float16)aF0[j]; a0[j + 4] = (_Float16)aF1[j];
            a1[j] = (_Float16)aF2[j]; a1[j + 4] = (_Float16)aF3[j];
        }
        dq16(bR, b0, b1);
        *(v8h*)(sA + 0) = a0;  *(v8h*)(sA + 8) = a1;   // ds_store_b128
        *(v8h*)(sB + 0) = b0;  *(v8h*)(sB + 8) = b1;
    }

    const int niter = K / BK;   // 128

    for (int it = 0; it < niter; ++it) {
        __syncthreads();   // chunk `it` now visible in buffer it&1

        // ---- issue global loads for chunk it+1 (overlap with WMMAs) --------
        const bool more = (it + 1) < niter;
        if (more) {
            const int kn = (it + 1) * BK;
            aF0 = *(const v4f*)(gA + kn + 0);
            aF1 = *(const v4f*)(gA + kn + 4);
            aF2 = *(const v4f*)(gA + kn + 8);
            aF3 = *(const v4f*)(gA + kn + 12);
            bR  = *(const int4*)(gB + kn);
            if (it + 2 < niter) {                      // global_prefetch_b8
                __builtin_prefetch(gA + kn + BK, 0, 0);
                __builtin_prefetch(gB + kn + BK, 0, 0);
            }
        }

        // ---- load ALL fragments up front, then 8 WMMAs ---------------------
        const _Float16* Ab = As[it & 1];
        const _Float16* Bb = Bs[it & 1];

        union F { v16h v; v8h h[2]; };
        F bfrag[2], afrag[4];
#pragma unroll
        for (int nt = 0; nt < 2; ++nt) {
            // B 32x16: lane = col n, 16 contiguous k at hi*16
            const _Float16* p = Bb + (wn * 32 + nt * 16 + lm) * LDH + hi * 16;
            bfrag[nt].h[0] = *(const v8h*)(p);         // ds_load_b128
            bfrag[nt].h[1] = *(const v8h*)(p + 8);
        }
#pragma unroll
        for (int mt = 0; mt < 4; ++mt) {
            // A 16x32: lane = row m, 16 contiguous k at hi*16
            const _Float16* p = Ab + (wm * 64 + mt * 16 + lm) * LDH + hi * 16;
            afrag[mt].h[0] = *(const v8h*)(p);
            afrag[mt].h[1] = *(const v8h*)(p + 8);
        }
#pragma unroll
        for (int mt = 0; mt < 4; ++mt)
#pragma unroll
            for (int nt = 0; nt < 2; ++nt)
                acc[mt][nt] = __builtin_amdgcn_wmma_f32_16x16x32_f16(
                    false, afrag[mt].v, false, bfrag[nt].v,
                    (short)0, acc[mt][nt], false, false);

        // ---- convert + store chunk it+1 into the other buffer --------------
        if (more) {
            _Float16* sA = &As[(it + 1) & 1][grow * LDH + gcol];
            _Float16* sB = &Bs[(it + 1) & 1][grow * LDH + gcol];
            v8h a0, a1, b0, b1;
#pragma unroll
            for (int j = 0; j < 4; ++j) {
                a0[j] = (_Float16)aF0[j]; a0[j + 4] = (_Float16)aF1[j];
                a1[j] = (_Float16)aF2[j]; a1[j + 4] = (_Float16)aF3[j];
            }
            dq16(bR, b0, b1);
            *(v8h*)(sA + 0) = a0;  *(v8h*)(sA + 8) = a1;
            *(v8h*)(sB + 0) = b0;  *(v8h*)(sB + 8) = b1;
        }
    }

    // ---- epilogue: per-output-channel scale, fp32 store --------------------
#pragma unroll
    for (int nt = 0; nt < 2; ++nt) {
        const int ncol = n0 + wn * 32 + nt * 16 + lm;
        const float scl = __bfloat162float(S[ncol]);
#pragma unroll
        for (int mt = 0; mt < 4; ++mt) {
#pragma unroll
            for (int r = 0; r < 8; ++r) {
                const int mrow = m0 + wm * 64 + mt * 16 + hi * 8 + r;
                Out[(size_t)mrow * N + ncol] = acc[mt][nt][r] * scl;
            }
        }
    }
}

extern "C" void kernel_launch(void* const* d_in, const int* in_sizes, int n_in,
                              void* d_out, int out_size, void* d_ws, size_t ws_size,
                              hipStream_t stream) {
    const float*          x = (const float*)d_in[0];          // [B,S,IN] fp32
    const int8_t*         w = (const int8_t*)d_in[1];         // [OUT,IN] int8
    const __hip_bfloat16* s = (const __hip_bfloat16*)d_in[2]; // [OUT] bf16
    float*              out = (float*)d_out;                  // [B,S,OUT] fp32

    const int M = 4 * 2048;   // B*S
    const int K = 4096;       // IN
    const int N = 11008;      // OUT

    dim3 grid(N / BN, M / BM);   // 86 x 64 workgroups, all tiles full
    woq_int8_gemm_wmma<<<grid, 256, 0, stream>>>(x, w, s, out, M, N, K);
}
